// SimpleDetector_24455543783494
// MI455X (gfx1250) — matrix operand order repaired and checked by reading, not compiled, not run on hardware
//
#include <hip/hip_runtime.h>
#include <hip/hip_bf16.h>

typedef __attribute__((ext_vector_type(16))) _Float16 v16h;
typedef __attribute__((ext_vector_type(8)))  _Float16 v8h;
typedef __attribute__((ext_vector_type(4)))  _Float16 v4h;
typedef __attribute__((ext_vector_type(8)))  float    v8f;
typedef __attribute__((ext_vector_type(4)))  unsigned int v4u;
typedef __attribute__((ext_vector_type(8)))  int      v8i;
typedef __attribute__((ext_vector_type(4)))  int      v4i;

#define TPB    256
#define WAVES  8
#define TILE_E 16
#define FEAT   128
#define LDW    136                     // padded LDS row stride in halves (272B)
#define W_LDS_BYTES (FEAT * LDW * 2)   // 34816 B per weight tile in LDS

// Build a v16h from two 16-byte LDS chunks (ds_load_b128 x2).
__device__ __forceinline__ v16h load_frag2(const _Float16* p0, const _Float16* p1) {
    v8h lo = *(const v8h*)p0;
    v8h hi = *(const v8h*)p1;
    v16h r;
#pragma unroll
    for (int i = 0; i < 8; ++i) { r[i] = lo[i]; r[i + 8] = hi[i]; }
    return r;
}

// Order-preserving float->uint encoding for atomic min.
__device__ __forceinline__ unsigned int fenc(float f) {
    unsigned int u = __float_as_uint(f);
    return (u & 0x80000000u) ? ~u : (u | 0x80000000u);
}

// TDM: load a dense f16 [128][128] tile from global into LDS with row padding
// 128 halves data + 8 halves pad = LDW stride. Descriptor per CDNA5 ISA D#.
__device__ __forceinline__ void tdm_load_wtile(const _Float16* gsrc, unsigned lds_off) {
    unsigned long long ga = (unsigned long long)(const void*)gsrc;
    v4u g0;
    g0[0] = 1u;                                       // count=1, user descriptor
    g0[1] = lds_off;                                  // lds_addr (bytes)
    g0[2] = (unsigned)(ga & 0xFFFFFFFFu);             // global_addr[31:0]
    g0[3] = (unsigned)((ga >> 32) & 0x01FFFFFFu)      // global_addr[56:32]
          | (2u << 30);                               // type = 2 ("image")
    v8i g1;
    g1[0] = (int)((1u << 16)      // data_size = 1 -> 2 bytes
                | (1u << 20)      // pad_enable
                | (5u << 22)      // pad_interval: 2^(5+1) = 64 DWORDs (=128 halves)
                | (3u << 25));    // pad_amount: 4 DWORDs (=8 halves) -> LDW=136
    g1[1] = (int)(128u << 16);    // tensor_dim0[15:0] = 128 (bits 79:48)
    g1[2] = (int)(128u << 16);    // tensor_dim1[15:0] = 128 (bits 111:80)
    g1[3] = (int)(128u << 16);    // tile_dim0 = 128 (bits 127:112)
    g1[4] = 128;                  // tile_dim1 = 128 (bits 143:128), tile_dim2 = 0
    g1[5] = 128;                  // tensor_dim0_stride = 128 (bits 207:160)
    g1[6] = 0;
    g1[7] = 0;
    v4i gz4 = {0, 0, 0, 0};       // groups 2/3 unused (2D tensor)
    v8i gz8 = {0, 0, 0, 0, 0, 0, 0, 0};
    __builtin_amdgcn_tensor_load_to_lds(g0, g1, gz4, gz4, gz8, 0);
}

__global__ void prep_weights_kernel(const float* __restrict__ W1,
                                    const float* __restrict__ W2,
                                    _Float16* __restrict__ w1t,
                                    _Float16* __restrict__ w2t) {
    int i = blockIdx.x * blockDim.x + threadIdx.x;
    if (i < FEAT * FEAT) {
        int n = i >> 7, k = i & (FEAT - 1);
        // transposed: [n][k]
        w1t[i] = (_Float16)W1[k * FEAT + n];
        w2t[i] = (_Float16)W2[k * FEAT + n];
    }
}

__global__ void init_out_kernel(unsigned int* __restrict__ out, int n) {
    int i = blockIdx.x * blockDim.x + threadIdx.x;
    if (i < n) out[i] = 0xFFFFFFFFu;   // sentinel: "no edge touched this segment"
}

__global__ void finalize_kernel(unsigned int* __restrict__ out, int n) {
    int i = blockIdx.x * blockDim.x + threadIdx.x;
    if (i < n) {
        unsigned int u = out[i];
        float r;
        if (u == 0xFFFFFFFFu) {
            r = 0.0f;                   // empty segment -> 0
        } else {
            unsigned int bits = (u & 0x80000000u) ? (u & 0x7FFFFFFFu) : ~u;
            r = __uint_as_float(bits);
        }
        ((float*)out)[i] = r;
    }
}

__global__ __launch_bounds__(TPB)
void edge_mlp_kernel(const float* __restrict__ zs,
                     const int*   __restrict__ src,
                     const int*   __restrict__ dst,
                     const int*   __restrict__ idx,
                     const float* __restrict__ W1, const float* __restrict__ b1,
                     const float* __restrict__ W2, const float* __restrict__ b2,
                     const float* __restrict__ W3, const float* __restrict__ b3,
                     const _Float16* __restrict__ wsW1t,
                     const _Float16* __restrict__ wsW2t,
                     int use_ws,
                     unsigned int* __restrict__ outbits,
                     int n_edges) {
    extern __shared__ char smem[];
    _Float16* w1t  = (_Float16*)smem;                 // 128 x LDW halves
    _Float16* w2t  = w1t + FEAT * LDW;                // 128 x LDW halves
    float*    w3s  = (float*)(w2t + FEAT * LDW);      // 128 f32
    float*    b1s  = w3s + FEAT;                      // 128 f32
    float*    b2s  = b1s + FEAT;                      // 128 f32
    _Float16* tiles = (_Float16*)(b2s + FEAT);        // per wave: 2 * 16 * LDW halves

    const int t = threadIdx.x;

    // ---- stage weights into LDS ----
    if (use_ws) {
        // Tensor Data Mover: one wave issues two 2D tile descriptors with
        // hardware row padding (128 halves + 8 pad = LDW), waits TENSORcnt.
        if (t == 0) {
            unsigned dynbase = (unsigned)__builtin_amdgcn_groupstaticsize();
            tdm_load_wtile(wsW1t, dynbase);
            tdm_load_wtile(wsW2t, dynbase + W_LDS_BYTES);
            __builtin_amdgcn_s_wait_tensorcnt(0);
        }
    } else {
        for (int i = t; i < FEAT * FEAT; i += TPB) {
            int n = i >> 7, k = i & (FEAT - 1);
            w1t[n * LDW + k] = (_Float16)W1[k * FEAT + n];
            w2t[n * LDW + k] = (_Float16)W2[k * FEAT + n];
        }
    }
    if (t < FEAT) { w3s[t] = W3[t]; b1s[t] = b1[t]; b2s[t] = b2[t]; }
    __syncthreads();

    const int wave = t >> 5;
    const int lane = t & 31;
    const int row  = lane & 15;   // M row of A / N col of B-C-D
    const int hh   = lane >> 4;   // lane half

    _Float16* xt = tiles + wave * 2 * TILE_E * LDW;
    _Float16* ht = xt + TILE_E * LDW;

    const int e0 = (int)blockIdx.x * (WAVES * TILE_E) + wave * TILE_E;

    // ---- gather + f16 convert: 16 edges x 128 concat feats ----
    // lanes 0..15 fetch src feats [lane*4..+4), lanes 16..31 dst feats.
    for (int e = 0; e < TILE_E; ++e) {
        int eg  = e0 + e;
        int egc = eg < n_edges ? eg : n_edges - 1;
        int s = src[egc], d = dst[egc];
        const float* base = (lane < 16) ? (zs + (size_t)s * 64 + lane * 4)
                                        : (zs + (size_t)d * 64 + (lane - 16) * 4);
        float4 v = *(const float4*)base;          // global_load_b128
        v4h hv;
        hv[0] = (_Float16)v.x; hv[1] = (_Float16)v.y;
        hv[2] = (_Float16)v.z; hv[3] = (_Float16)v.w;
        *(v4h*)(xt + e * LDW + lane * 4) = hv;    // ds_store_b64
    }
    asm volatile("s_wait_dscnt 0" ::: "memory");

    // ---- layer 1: h1 = relu(x @ W1 + b1), via 8 nb x 4 kb WMMA ----
    v16h afrag[4];
#pragma unroll
    for (int kb = 0; kb < 4; ++kb) {
        const _Float16* base = xt + row * LDW + kb * 32 + hh * 8;
        afrag[kb] = load_frag2(base, base + 16);
    }
#pragma unroll
    for (int nb = 0; nb < 8; ++nb) {
        float bias = b1s[nb * 16 + row];
        v8f acc;
#pragma unroll
        for (int r = 0; r < 8; ++r) acc[r] = bias;
#pragma unroll
        for (int kb = 0; kb < 4; ++kb) {
            const _Float16* wb = w1t + (nb * 16 + row) * LDW + kb * 32 + hh * 8;
            v16h bfrag = load_frag2(wb, wb + 16);
            acc = __builtin_amdgcn_wmma_f32_16x16x32_f16(
                false, afrag[kb], false, bfrag, (short)0, acc, false, false);
        }
#pragma unroll
        for (int r = 0; r < 8; ++r) {
            ht[(r + 8 * hh) * LDW + nb * 16 + row] = (_Float16)fmaxf(acc[r], 0.0f);
        }
    }
    asm volatile("s_wait_dscnt 0" ::: "memory");

    // ---- layer 2: h2 = relu(h1 @ W2 + b2) -> write back into xt ----
#pragma unroll
    for (int kb = 0; kb < 4; ++kb) {
        const _Float16* base = ht + row * LDW + kb * 32 + hh * 8;
        afrag[kb] = load_frag2(base, base + 16);
    }
#pragma unroll
    for (int nb = 0; nb < 8; ++nb) {
        float bias = b2s[nb * 16 + row];
        v8f acc;
#pragma unroll
        for (int r = 0; r < 8; ++r) acc[r] = bias;
#pragma unroll
        for (int kb = 0; kb < 4; ++kb) {
            const _Float16* wb = w2t + (nb * 16 + row) * LDW + kb * 32 + hh * 8;
            v16h bfrag = load_frag2(wb, wb + 16);
            acc = __builtin_amdgcn_wmma_f32_16x16x32_f16(
                false, afrag[kb], false, bfrag, (short)0, acc, false, false);
        }
#pragma unroll
        for (int r = 0; r < 8; ++r) {
            xt[(r + 8 * hh) * LDW + nb * 16 + row] = (_Float16)fmaxf(acc[r], 0.0f);
        }
    }
    asm volatile("s_wait_dscnt 0" ::: "memory");

    // ---- final: pred = h2 @ W3 + b3; split 128-dot across lane halves ----
    float sum = 0.0f;
    const _Float16* yrow = xt + row * LDW + hh * 64;
    const float*    w3p  = w3s + hh * 64;
#pragma unroll
    for (int c = 0; c < 8; ++c) {
        v8h    yv = *(const v8h*)(yrow + c * 8);          // ds_load_b128
        float4 wa = *(const float4*)(w3p + c * 8);
        float4 wb = *(const float4*)(w3p + c * 8 + 4);
        sum += (float)yv[0] * wa.x + (float)yv[1] * wa.y +
               (float)yv[2] * wa.z + (float)yv[3] * wa.w +
               (float)yv[4] * wb.x + (float)yv[5] * wb.y +
               (float)yv[6] * wb.z + (float)yv[7] * wb.w;
    }
    sum += __shfl_xor(sum, 16, 32);

    int eg = e0 + row;
    if (hh == 0 && eg < n_edges) {
        float pred = sum + b3[0];
        atomicMin(&outbits[idx[eg]], fenc(pred));
    }
}

extern "C" void kernel_launch(void* const* d_in, const int* in_sizes, int n_in,
                              void* d_out, int out_size, void* d_ws, size_t ws_size,
                              hipStream_t stream) {
    const float* zs  = (const float*)d_in[0];
    const int*   src = (const int*)d_in[1];
    const int*   dst = (const int*)d_in[2];
    const int*   idx = (const int*)d_in[3];
    const float* W1  = (const float*)d_in[4];
    const float* b1  = (const float*)d_in[5];
    const float* W2  = (const float*)d_in[6];
    const float* b2  = (const float*)d_in[7];
    const float* W3  = (const float*)d_in[8];
    const float* b3  = (const float*)d_in[9];

    const int E = in_sizes[1];       // number of edges
    const int P = out_size;          // number of procs / segments

    const size_t wbytes = (size_t)FEAT * FEAT * sizeof(_Float16);
    int use_ws = (ws_size >= 2 * wbytes) ? 1 : 0;
    _Float16* wsW1t = (_Float16*)d_ws;
    _Float16* wsW2t = wsW1t + FEAT * FEAT;

    if (use_ws) {
        prep_weights_kernel<<<(FEAT * FEAT + 255) / 256, 256, 0, stream>>>(W1, W2, wsW1t, wsW2t);
    }

    unsigned int* outbits = (unsigned int*)d_out;
    init_out_kernel<<<(P + 255) / 256, 256, 0, stream>>>(outbits, P);

    const size_t shmem =
        2 * (size_t)FEAT * LDW * sizeof(_Float16) +   // W1t, W2t
        3 * (size_t)FEAT * sizeof(float) +            // W3, b1, b2
        (size_t)WAVES * 2 * TILE_E * LDW * sizeof(_Float16);

    (void)hipFuncSetAttribute((const void*)edge_mlp_kernel,
                              hipFuncAttributeMaxDynamicSharedMemorySize,
                              (int)shmem);

    const int edges_per_block = WAVES * TILE_E;  // 128
    int nblocks = (E + edges_per_block - 1) / edges_per_block;
    edge_mlp_kernel<<<nblocks, TPB, shmem, stream>>>(
        zs, src, dst, idx, W1, b1, W2, b2, W3, b3,
        wsW1t, wsW2t, use_ws, outbits, E);

    finalize_kernel<<<(P + 255) / 256, 256, 0, stream>>>(outbits, P);
}
